// QuestionAnwserLogits_30296699306624
// MI455X (gfx1250) — matrix operand order repaired and checked by reading, not compiled, not run on hardware
//
#include <hip/hip_runtime.h>

// ---------------------------------------------------------------------------
// Types for CDNA5 WMMA (gfx1250, wave32)
// ---------------------------------------------------------------------------
typedef __attribute__((ext_vector_type(16))) __bf16 bf16x16;
typedef __attribute__((ext_vector_type(8)))  float  v8f;

union ABFrag {
    bf16x16 v;
    uint4   q[2];
};

__device__ __forceinline__ unsigned int bf16_rne(float x) {
    unsigned int u = __float_as_uint(x);
    u += 0x7fffu + ((u >> 16) & 1u);   // round-to-nearest-even
    return u >> 16;
}
__device__ __forceinline__ unsigned int pack_bf16x2(float lo, float hi) {
    return bf16_rne(lo) | (bf16_rne(hi) << 16);
}
// sum across the 16-lane group that shares one matrix row (xor masks 1,2,4,8)
__device__ __forceinline__ float rsum16(float v) {
    #pragma unroll
    for (int m = 1; m < 16; m <<= 1) v += __shfl_xor(v, m, 32);
    return v;
}

#define BT   16384     // B*T
#define TT   2048      // T
#define HH   1024      // H
#define MASK_NEG 1e30f

// ---------------------------------------------------------------------------
// K0: convert end_w0[0:H, :] (f32 row-major [1024][1024]) to bf16 in a
// fragment-native layout wsW[kb][n][kk]  (kb = k/32, kk = k%32).
// Lane (n, half h) then reads bytes [16h .. 16h+15] and [32+16h .. 32+16h+15] of a contiguous 64B run.
// ---------------------------------------------------------------------------
__global__ __launch_bounds__(256) void convW_kernel(const float* __restrict__ w0,
                                                    unsigned short* __restrict__ wsW) {
    int idx = blockIdx.x * 256 + threadIdx.x;      // 0 .. 1M-1
    int k = idx >> 10;
    int n = idx & 1023;
    float v = w0[idx];                             // w0[k*1024 + n], k < 1024
    wsW[((k >> 5) << 15) + (n << 5) + (k & 31)] = (unsigned short)bf16_rne(v);
}

// ---------------------------------------------------------------------------
// K1: start_logits (masked) -> out[0:BT].  One wave per row.
// ---------------------------------------------------------------------------
__global__ __launch_bounds__(256) void start_logits_kernel(
    const float* __restrict__ inputs, const float* __restrict__ pmask,
    const float* __restrict__ sw, const float* __restrict__ sb,
    float* __restrict__ out) {
    int row  = blockIdx.x * 8 + (threadIdx.x >> 5);
    int lane = threadIdx.x & 31;
    const float4* x = (const float4*)(inputs + (size_t)row * HH);
    const float4* w = (const float4*)sw;
    float acc = 0.f;
    #pragma unroll
    for (int j = 0; j < 8; ++j) {
        float4 a = x[j * 32 + lane];
        float4 b = w[j * 32 + lane];
        acc += a.x * b.x + a.y * b.y + a.z * b.z + a.w * b.w;
    }
    #pragma unroll
    for (int m = 1; m < 32; m <<= 1) acc += __shfl_xor(acc, m, 32);
    if (lane == 0) {
        float pm = pmask[row];
        float v  = acc + sb[0];
        out[row] = v * (1.0f - pm) - MASK_NEG * pm;
    }
}

// ---------------------------------------------------------------------------
// K2: bias2[b,n] = start_feat[b,:] @ end_w0[H:, n] + end_b0[n]
// ---------------------------------------------------------------------------
__global__ __launch_bounds__(256) void bias2_kernel(
    const float* __restrict__ inputs, const int* __restrict__ spos,
    const float* __restrict__ w0, const float* __restrict__ b0,
    float* __restrict__ bias2) {
    int b = blockIdx.x >> 2;
    int h = (blockIdx.x & 3) * 256 + threadIdx.x;
    const float* sf = inputs + ((size_t)b * TT + spos[b]) * HH;
    const float* w  = w0 + (size_t)HH * HH + h;     // rows H..2H-1
    float acc = b0[h];
    for (int f = 0; f < HH; ++f) acc += sf[f] * w[(size_t)f * HH];
    bias2[b * HH + h] = acc;
}

// ---------------------------------------------------------------------------
// K3: per-batch softmax over masked start logits -> p[B,T]
// ---------------------------------------------------------------------------
__global__ __launch_bounds__(256) void softmax_kernel(const float* __restrict__ sl,
                                                      float* __restrict__ p) {
    __shared__ float red[256];
    int b = blockIdx.x, tid = threadIdx.x;
    const float* x = sl + b * TT;
    float mx = -3.0e38f;
    for (int i = tid; i < TT; i += 256) mx = fmaxf(mx, x[i]);
    red[tid] = mx; __syncthreads();
    for (int s = 128; s > 0; s >>= 1) { if (tid < s) red[tid] = fmaxf(red[tid], red[tid + s]); __syncthreads(); }
    mx = red[0]; __syncthreads();
    float sum = 0.f;
    for (int i = tid; i < TT; i += 256) { float e = expf(x[i] - mx); p[b * TT + i] = e; sum += e; }
    red[tid] = sum; __syncthreads();
    for (int s = 128; s > 0; s >>= 1) { if (tid < s) red[tid] += red[tid + s]; __syncthreads(); }
    float inv = 1.0f / red[0];
    for (int i = tid; i < TT; i += 256) p[b * TT + i] *= inv;
}

// ---------------------------------------------------------------------------
// K4: start_feature[b,h] = sum_t p[b,t] * inputs[b,t,h]
// ---------------------------------------------------------------------------
__global__ __launch_bounds__(256) void startfeat_kernel(
    const float* __restrict__ inputs, const float* __restrict__ p,
    float* __restrict__ sf) {
    int b = blockIdx.x >> 2;
    int h = (blockIdx.x & 3) * 256 + threadIdx.x;
    const float* xp = inputs + (size_t)b * TT * HH + h;
    const float* pp = p + b * TT;
    float acc = 0.f;
    for (int t = 0; t < TT; ++t) acc += pp[t] * xp[(size_t)t * HH];
    sf[b * HH + h] = acc;
}

// ---------------------------------------------------------------------------
// K5: ans = tanh([start_feature | cls_feat] @ ans_w0 + ans_b0)
// ---------------------------------------------------------------------------
__global__ __launch_bounds__(256) void ans_kernel(
    const float* __restrict__ inputs, const int* __restrict__ cidx,
    const float* __restrict__ sf, const float* __restrict__ aw0,
    const float* __restrict__ ab0, float* __restrict__ ans) {
    int b = blockIdx.x >> 2;
    int h = (blockIdx.x & 3) * 256 + threadIdx.x;
    const float* cf = inputs + ((size_t)b * TT + cidx[b]) * HH;
    const float* s  = sf + b * HH;
    float acc = ab0[h];
    for (int f = 0; f < HH; ++f) acc += s[f]  * aw0[(size_t)f * HH + h];
    for (int f = 0; f < HH; ++f) acc += cf[f] * aw0[(size_t)(HH + f) * HH + h];
    ans[b * HH + h] = tanhf(acc);
}

// ---------------------------------------------------------------------------
// K6: cls_logits[b] = ans[b,:] @ ans_w1
// ---------------------------------------------------------------------------
__global__ __launch_bounds__(256) void cls_kernel(const float* __restrict__ ans,
                                                  const float* __restrict__ aw1,
                                                  float* __restrict__ out) {
    __shared__ float red[256];
    int b = blockIdx.x, tid = threadIdx.x;
    float acc = 0.f;
    for (int h = tid; h < HH; h += 256) acc += ans[b * HH + h] * aw1[h];
    red[tid] = acc; __syncthreads();
    for (int s = 128; s > 0; s >>= 1) { if (tid < s) red[tid] += red[tid + s]; __syncthreads(); }
    if (tid == 0) out[b] = red[0];
}

// ---------------------------------------------------------------------------
// K7: WMMA GEMM  C[row,n] = inputs[row,:] @ W0a[:,n], fused:
//     tanh(C + bias2) -> LayerNorm over n -> dot end_w1 -> mask -> end logits
// Workgroup: 256 threads (8 waves), 32 rows x 1024 cols; wave w owns cols
// [w*128, w*128+128), two 16-row M blocks, 2x8 accumulator tiles.
// ---------------------------------------------------------------------------
__global__ __launch_bounds__(256) void end_gemm_kernel(
    const float* __restrict__ inputs, const float* __restrict__ pmask,
    const unsigned short* __restrict__ wsW, const float* __restrict__ bias2,
    const float* __restrict__ gamma, const float* __restrict__ beta,
    const float* __restrict__ w1, const float* __restrict__ b1,
    float* __restrict__ out_end) {
    __shared__ __align__(16) unsigned int lds_a[32 * 16];   // 32 rows x 32 bf16 (64B/row)
    __shared__ float row_s[32], row_q[32], row_e[32];

    const int tid     = threadIdx.x;
    const int lane    = tid & 31;
    const int wave    = tid >> 5;
    const int rowbase = blockIdx.x * 32;          // 32 | 2048 -> single batch
    const int b       = rowbase >> 11;
    const int m       = lane & 15;                // row-in-tile / col-in-tile
    const int h       = lane >> 4;                // lane half
    const int nbase   = wave * 128;

    v8f c0[8] = {};                               // rows rowbase+0..15
    v8f c1[8] = {};                               // rows rowbase+16..31

    // A staging: thread loads one float4 of one row per k-step
    const int ar = tid >> 3;                      // 0..31
    const int aj = tid & 7;                       // float4 index 0..7
    const uint4* wb = (const uint4*)wsW;

    for (int kb = 0; kb < 32; ++kb) {
        __syncthreads();
        const float4 av = *(const float4*)(inputs + (size_t)(rowbase + ar) * HH + kb * 32 + aj * 4);
        lds_a[ar * 16 + aj * 2]     = pack_bf16x2(av.x, av.y);
        lds_a[ar * 16 + aj * 2 + 1] = pack_bf16x2(av.z, av.w);
        __syncthreads();

        // A fragments (ISA 16-bit A 16x32 layout): two 16B runs per lane
        ABFrag a0, a1;
        {
            const unsigned int* p0 = &lds_a[m * 16 + 4 * h];
            a0.q[0] = *(const uint4*)(p0);
            a0.q[1] = *(const uint4*)(p0 + 8);
            const unsigned int* p1 = &lds_a[(m + 16) * 16 + 4 * h];
            a1.q[0] = *(const uint4*)(p1);
            a1.q[1] = *(const uint4*)(p1 + 8);
        }
        #pragma unroll
        for (int t = 0; t < 8; ++t) {
            const int n = nbase + t * 16 + m;
            ABFrag bf;
            const int bi = (kb * 1024 + n) * 4 + h;       // uint4 index into wsW
            bf.q[0] = wb[bi];
            bf.q[1] = wb[bi + 2];
            c0[t] = __builtin_amdgcn_wmma_f32_16x16x32_bf16(
                        false, a0.v, false, bf.v, (short)0, c0[t], false, false);
            c1[t] = __builtin_amdgcn_wmma_f32_16x16x32_bf16(
                        false, a1.v, false, bf.v, (short)0, c1[t], false, false);
        }
    }

    // ---- fused epilogue ----
    __syncthreads();
    if (tid < 32) { row_s[tid] = 0.f; row_q[tid] = 0.f; row_e[tid] = 0.f; }

    float bias_t[8], g_t[8], be_t[8], w1_t[8];
    #pragma unroll
    for (int t = 0; t < 8; ++t) {
        const int n = nbase + t * 16 + m;
        bias_t[t] = bias2[b * HH + n];
        g_t[t]    = gamma[n];
        be_t[t]   = beta[n];
        w1_t[t]   = w1[n];
    }
    __syncthreads();

    // pass 1: tanh(acc + bias), accumulate per-row sum / sumsq
    #pragma unroll
    for (int blk = 0; blk < 2; ++blk) {
        v8f* C = blk ? c1 : c0;
        #pragma unroll
        for (int r = 0; r < 8; ++r) {
            float s = 0.f, qq = 0.f;
            #pragma unroll
            for (int t = 0; t < 8; ++t) {
                float v = tanhf(C[t][r] + bias_t[t]);
                C[t][r] = v;
                s += v; qq += v * v;
            }
            s  = rsum16(s);
            qq = rsum16(qq);
            if (m == 0) {
                const int row = blk * 16 + 8 * h + r;     // row within block of 32
                atomicAdd(&row_s[row], s);
                atomicAdd(&row_q[row], qq);
            }
        }
    }
    __syncthreads();

    // pass 2: LayerNorm + dot with end_w1
    #pragma unroll
    for (int blk = 0; blk < 2; ++blk) {
        v8f* C = blk ? c1 : c0;
        #pragma unroll
        for (int r = 0; r < 8; ++r) {
            const int row = blk * 16 + 8 * h + r;
            const float mu   = row_s[row] * (1.0f / (float)HH);
            const float var  = row_q[row] * (1.0f / (float)HH) - mu * mu;
            const float rstd = rsqrtf(var + 1e-12f);
            float d = 0.f;
            #pragma unroll
            for (int t = 0; t < 8; ++t)
                d += ((C[t][r] - mu) * rstd * g_t[t] + be_t[t]) * w1_t[t];
            d = rsum16(d);
            if (m == 0) atomicAdd(&row_e[row], d);
        }
    }
    __syncthreads();

    if (tid < 32) {
        const int row = rowbase + tid;
        const float pm = pmask[row];
        const float v  = row_e[tid] + b1[0];
        out_end[row] = v * (1.0f - pm) - MASK_NEG * pm;
    }
}

// ---------------------------------------------------------------------------
// Host launcher
// ---------------------------------------------------------------------------
extern "C" void kernel_launch(void* const* d_in, const int* in_sizes, int n_in,
                              void* d_out, int out_size, void* d_ws, size_t ws_size,
                              hipStream_t stream) {
    const float* inputs  = (const float*)d_in[0];
    const float* pmask   = (const float*)d_in[1];
    const int*   cidx    = (const int*)  d_in[2];
    const int*   spos    = (const int*)  d_in[3];
    const float* start_w = (const float*)d_in[4];
    const float* start_b = (const float*)d_in[5];
    const float* end_w0  = (const float*)d_in[6];
    const float* end_b0  = (const float*)d_in[7];
    const float* ln_g    = (const float*)d_in[8];
    const float* ln_b    = (const float*)d_in[9];
    const float* end_w1  = (const float*)d_in[10];
    const float* end_b1  = (const float*)d_in[11];
    const float* ans_w0  = (const float*)d_in[12];
    const float* ans_b0  = (const float*)d_in[13];
    const float* ans_w1  = (const float*)d_in[14];

    float* out = (float*)d_out;
    char*  ws  = (char*)d_ws;
    unsigned short* wsW  = (unsigned short*)(ws);                    // 2 MB
    float* ws_p   = (float*)(ws + (2u << 20));                       // 64 KB
    float* ws_sf  = (float*)(ws + (2u << 20) + 65536u);              // 32 KB
    float* ws_b2  = (float*)(ws + (2u << 20) + 65536u + 32768u);     // 32 KB
    float* ws_ans = (float*)(ws + (2u << 20) + 65536u + 65536u);     // 32 KB

    float* out_start = out;            // [B*T]
    float* out_end   = out + BT;       // [B*T]
    float* out_cls   = out + 2 * BT;   // [B]

    convW_kernel       <<<4096, 256, 0, stream>>>(end_w0, wsW);
    start_logits_kernel<<<BT / 8, 256, 0, stream>>>(inputs, pmask, start_w, start_b, out_start);
    bias2_kernel       <<<32, 256, 0, stream>>>(inputs, spos, end_w0, end_b0, ws_b2);
    softmax_kernel     <<<8, 256, 0, stream>>>(out_start, ws_p);
    startfeat_kernel   <<<32, 256, 0, stream>>>(inputs, ws_p, ws_sf);
    ans_kernel         <<<32, 256, 0, stream>>>(inputs, cidx, ws_sf, ans_w0, ans_b0, ws_ans);
    cls_kernel         <<<8, 256, 0, stream>>>(ws_ans, ans_w1, out_cls);
    end_gemm_kernel    <<<BT / 32, 256, 0, stream>>>(inputs, pmask, wsW, ws_b2,
                                                     ln_g, ln_b, end_w1, end_b1, out_end);
}
